// CasualSelfAttention_17987323036467
// MI455X (gfx1250) — compile-verified, matrix operand order checked by reading, not run on hardware
//
#include <hip/hip_runtime.h>
#include <hip/hip_bf16.h>

typedef __attribute__((ext_vector_type(16))) _Float16 v16h;
typedef __attribute__((ext_vector_type(8)))  _Float16 v8h;
typedef __attribute__((ext_vector_type(8)))  float    v8f;

#define B_  4
#define T_  2048
#define C_  1024
#define H_  16
#define D_  64

// Build a 16x32 f16 A/B fragment (v16h) from two contiguous 16-byte runs.
// CDNA5 WMMA layout: lane (lr,kg) holds k in [kg*8, kg*8+8) in VGPRs 0-3 and
// k in [16+kg*8, 16+kg*8+8) in VGPRs 4-7. (A rows and B columns pack the same.)
static __device__ __forceinline__ v16h load_frag(const _Float16* lo, const _Float16* hi) {
    v8h a = *(const v8h*)lo;
    v8h b = *(const v8h*)hi;
    v16h r;
#pragma unroll
    for (int i = 0; i < 8; ++i) { r[i] = a[i]; r[i + 8] = b[i]; }
    return r;
}

// Exchange with lane^16 (cross-half) via ds_swizzle: and=0x1f, or=0, xor=16.
static __device__ __forceinline__ float swz_xor16(float x) {
    int yi = __builtin_amdgcn_ds_swizzle(__builtin_bit_cast(int, x), 0x401F);
    return __builtin_bit_cast(float, yi);
}

// ---------------------------------------------------------------------------
// f32 -> f16 conversion
// ---------------------------------------------------------------------------
__global__ void f32_to_f16_kernel(const float* __restrict__ in,
                                  _Float16* __restrict__ out, long n) {
    long i = (long)blockIdx.x * blockDim.x + threadIdx.x;
    if (i < n) out[i] = (_Float16)in[i];
}

// ---------------------------------------------------------------------------
// Transpose + convert: Wt[n][k] = (f16) W[k][n],  W: [K,N] f32
// ---------------------------------------------------------------------------
__global__ __launch_bounds__(256)
void transpose_f32_to_f16_kernel(const float* __restrict__ in,
                                 _Float16* __restrict__ out, int K, int N) {
    __shared__ float tile[32][33];
    const int n0 = blockIdx.x * 32;
    const int k0 = blockIdx.y * 32;
    const int tx = threadIdx.x & 31;
    const int ty = threadIdx.x >> 5;  // 0..7
#pragma unroll
    for (int i = 0; i < 4; ++i) {
        const int k = ty + i * 8;
        tile[k][tx] = in[(size_t)(k0 + k) * N + n0 + tx];
    }
    __syncthreads();
#pragma unroll
    for (int i = 0; i < 4; ++i) {
        const int n = ty + i * 8;
        out[(size_t)(n0 + n) * K + k0 + tx] = (_Float16)tile[tx][n];
    }
}

// ---------------------------------------------------------------------------
// LDS-free WMMA GEMM, transposed accumulators:
//   computes C^T tiles via  D = Bt_tile (A-op) x A_rows (B-op)
//   -> C-layout: lane = output ROW m, VGPR index = output COLUMN n
//   -> epilogue is ONE contiguous v8h/v8f store per tile + vector bias add.
// Operands are L2-resident (192MB L2 vs ~24MB f16); both fragments are direct
// aligned global_load_b128 runs. Block tile 128x128: 8 waves (2 m-groups x
// 4 n-groups), each wave 64(m) x 32(n) = 2x4 WMMA accumulators, K step 32.
// ---------------------------------------------------------------------------
template <bool HALF_OUT>
__global__ __launch_bounds__(256)
void gemm_wmma_kernel(const _Float16* __restrict__ A,   // [M,K]
                      const _Float16* __restrict__ Bt,  // [N,K]
                      const float* __restrict__ bias,   // [N]
                      _Float16* __restrict__ Ch,
                      float* __restrict__ Cf,
                      int M, int N, int K) {
    const int lane = threadIdx.x & 31;
    const int wid  = threadIdx.x >> 5;
    const int lr   = lane & 15;
    const int kg   = lane >> 4;
    const int wm   = wid & 1;   // 0..1 : 64-row (m) group
    const int wn   = wid >> 1;  // 0..3 : 32-col (n) group

    const int bm0 = blockIdx.y * 128;
    const int bn0 = blockIdx.x * 128;

    const _Float16* nrow[2];  // WMMA A-operand: Bt rows (output columns)
    const _Float16* mrow[4];  // WMMA B-operand: A rows  (output rows)
#pragma unroll
    for (int nt = 0; nt < 2; ++nt)
        nrow[nt] = Bt + (size_t)(bn0 + wn * 32 + nt * 16 + lr) * K;
#pragma unroll
    for (int mt = 0; mt < 4; ++mt)
        mrow[mt] = A + (size_t)(bm0 + wm * 64 + mt * 16 + lr) * K;

    v8f acc[2][4];
#pragma unroll
    for (int nt = 0; nt < 2; ++nt)
#pragma unroll
        for (int mt = 0; mt < 4; ++mt) acc[nt][mt] = {};

    for (int k0 = 0; k0 < K; k0 += 32) {
        v16h nf[2], mf[4];
#pragma unroll
        for (int nt = 0; nt < 2; ++nt)
            nf[nt] = load_frag(nrow[nt] + k0 + kg * 8, nrow[nt] + k0 + 16 + kg * 8);
#pragma unroll
        for (int mt = 0; mt < 4; ++mt)
            mf[mt] = load_frag(mrow[mt] + k0 + kg * 8, mrow[mt] + k0 + 16 + kg * 8);

        if (k0 + 32 < K) {  // lowers to global_prefetch_b8
            __builtin_prefetch(nrow[0] + k0 + 32, 0, 1);
            __builtin_prefetch(nrow[1] + k0 + 32, 0, 1);
            __builtin_prefetch(mrow[0] + k0 + 32, 0, 1);
            __builtin_prefetch(mrow[2] + k0 + 32, 0, 1);
        }

#pragma unroll
        for (int nt = 0; nt < 2; ++nt)
#pragma unroll
            for (int mt = 0; mt < 4; ++mt)
                acc[nt][mt] = __builtin_amdgcn_wmma_f32_16x16x32_f16(
                    false, nf[nt], false, mf[mt], (short)0, acc[nt][mt], false, false);
    }

    // Epilogue: lane = output row, acc elements = 8 consecutive output columns.
#pragma unroll
    for (int nt = 0; nt < 2; ++nt) {
        const int ncol = bn0 + wn * 32 + nt * 16 + 8 * kg;
        const v8f bv = *(const v8f*)(bias + ncol);
#pragma unroll
        for (int mt = 0; mt < 4; ++mt) {
            const int m = bm0 + wm * 64 + mt * 16 + lr;
            if (HALF_OUT) {
                v8h o;
#pragma unroll
                for (int v = 0; v < 8; ++v) o[v] = (_Float16)(acc[nt][mt][v] + bv[v]);
                *(v8h*)(Ch + (size_t)m * N + ncol) = o;
            } else {
                v8f o;
#pragma unroll
                for (int v = 0; v < 8; ++v) o[v] = acc[nt][mt][v] + bv[v];
                *(v8f*)(Cf + (size_t)m * N + ncol) = o;
            }
        }
    }
}

// ---------------------------------------------------------------------------
// Extract V^T: vt[b][h][d][t] = qkv[(b*T+t)*3C + 2C + h*D + d]
// ---------------------------------------------------------------------------
__global__ __launch_bounds__(256)
void extract_vt_kernel(const _Float16* __restrict__ qkv,
                       _Float16* __restrict__ vt) {
    __shared__ __align__(16) _Float16 tile[64][72];  // [t][d], row = 144B
    const int blk    = blockIdx.x;
    const int tchunk = blk % (T_ / 64);
    const int bh     = blk / (T_ / 64);
    const int b = bh / H_, h = bh % H_;
    const int t0  = tchunk * 64;
    const int tid = threadIdx.x;

#pragma unroll
    for (int i = 0; i < 2; ++i) {
        const int chunk = i * 256 + tid;  // (t, dgroup)
        const int t  = chunk >> 3;
        const int dg = chunk & 7;
        *(v8h*)&tile[t][dg * 8] =
            *(const v8h*)(qkv + (size_t)(b * T_ + t0 + t) * (3 * C_) + 2 * C_ + h * D_ + dg * 8);
    }
    __syncthreads();
#pragma unroll
    for (int i = 0; i < 2; ++i) {
        const int chunk = i * 256 + tid;  // (d, tgroup)
        const int d  = chunk >> 3;
        const int tg = chunk & 7;
        v8h o;
#pragma unroll
        for (int u = 0; u < 8; ++u) o[u] = tile[tg * 8 + u][d];
        *(v8h*)(vt + ((size_t)(b * H_ + h) * D_ + d) * T_ + t0 + tg * 8) = o;
    }
}

// ---------------------------------------------------------------------------
// Flash attention, transposed orientation, fully LDS-free:
//   S^T = K @ Q^T      (same loads as Q@K^T, operands swapped)
//   C-layout of S^T: lane = query, VGPR index = key  ->  per-LANE softmax
//   P^T B-fragment assembles IN REGISTERS (lane (lr,kg) of S^T holds exactly
//   keys [8kg,8kg+8) of each 16-key tile = the B-fragment k-run pattern)
//   O^T = V^T @ P^T    (V^T A-fragment = contiguous vt row loads)
//   denominator accl += ones @ P^T (per-lane, no broadcast)
// One wave per 16-query tile; 64-key blocks; only last block is masked.
// exp base-2 with log2(e) folded into the Q scale.
// ---------------------------------------------------------------------------
__global__ __launch_bounds__(256)
void attn_kernel(const _Float16* __restrict__ qkv,
                 const _Float16* __restrict__ vt,
                 _Float16* __restrict__ y) {
    const int tid  = threadIdx.x;
    const int lane = tid & 31;
    const int wid  = tid >> 5;
    const int lr   = lane & 15;
    const int kg   = lane >> 4;

    const int gw = blockIdx.x * 8 + wid;
    const int qt = gw % (T_ / 16);
    const int bh = gw / (T_ / 16);
    const int b  = bh / H_;
    const int h  = bh % H_;
    const int q0 = qt * 16;
    const int q  = q0 + lr;  // this lane's query (column of S^T)

    const float QSCALE = 0.18033688f;  // log2(e) / sqrt(64)

    // Q^T B-fragments (lane holds Q row q, contiguous dim runs), pre-scaled
    v16h qa[2];
    {
        const _Float16* qp = qkv + (size_t)(b * T_ + q) * (3 * C_) + h * D_;
#pragma unroll
        for (int s = 0; s < 2; ++s) {
            qa[s] = load_frag(qp + s * 32 + kg * 8, qp + s * 32 + 16 + kg * 8);
#pragma unroll
            for (int i = 0; i < 16; ++i) qa[s][i] = qa[s][i] * (_Float16)QSCALE;
        }
    }
    // all-ones A fragment for the column-sum (denominator) WMMA
    v16h ones;
#pragma unroll
    for (int i = 0; i < 16; ++i) ones[i] = (_Float16)1.0f;

    // V^T row pointers for the 4 output d-tiles (A-fragment rows)
    const _Float16* vrow[4];
#pragma unroll
    for (int dt = 0; dt < 4; ++dt)
        vrow[dt] = vt + ((size_t)(b * H_ + h) * D_ + dt * 16 + lr) * T_;

    v8f acc[4], accl;
#pragma unroll
    for (int dt = 0; dt < 4; ++dt) acc[dt] = {};
    accl = {};
    float m = -1e30f;  // per-lane running max (this lane's query)

    const int nblocks = (q0 + 16 + 63) >> 6;  // causal key blocks of 64

    auto process_block = [&](int k0, bool masked) {
        // ---- S^T = K @ Q^T : four 16(key)x16(query) tiles ----
        v8f st[4];
#pragma unroll
        for (int nt = 0; nt < 4; ++nt) {
            st[nt] = {};
            const _Float16* kp =
                qkv + (size_t)(b * T_ + k0 + nt * 16 + lr) * (3 * C_) + C_ + h * D_;
#pragma unroll
            for (int s = 0; s < 2; ++s) {
                v16h kb = load_frag(kp + s * 32 + kg * 8, kp + s * 32 + 16 + kg * 8);
                st[nt] = __builtin_amdgcn_wmma_f32_16x16x32_f16(
                    false, kb, false, qa[s], (short)0, st[nt], false, false);
            }
        }
        if (masked) {  // only the diagonal block pays for compares
#pragma unroll
            for (int nt = 0; nt < 4; ++nt)
#pragma unroll
                for (int v = 0; v < 8; ++v) {
                    const int key = k0 + nt * 16 + v + 8 * kg;
                    if (key > q) st[nt][v] = -1e30f;
                }
        }

        // ---- per-lane online softmax (keys live in VGPR indices) ----
        float x = st[0][0];
#pragma unroll
        for (int nt = 0; nt < 4; ++nt)
#pragma unroll
            for (int v = 0; v < 8; ++v)
                if (nt | v) x = fmaxf(x, st[nt][v]);
        x = fmaxf(x, swz_xor16(x));  // other half holds the other 8 keys/tile
        const float mnew  = fmaxf(m, x);
        const float alpha = __builtin_amdgcn_exp2f(m - mnew);
        m = mnew;

        // P^T B-fragments assembled in registers (no LDS!)
        v16h pb01, pb23;
#pragma unroll
        for (int v = 0; v < 8; ++v) {
            pb01[v]     = (_Float16)__builtin_amdgcn_exp2f(st[0][v] - mnew);
            pb01[v + 8] = (_Float16)__builtin_amdgcn_exp2f(st[1][v] - mnew);
            pb23[v]     = (_Float16)__builtin_amdgcn_exp2f(st[2][v] - mnew);
            pb23[v + 8] = (_Float16)__builtin_amdgcn_exp2f(st[3][v] - mnew);
        }
#pragma unroll
        for (int dt = 0; dt < 4; ++dt)
#pragma unroll
            for (int v = 0; v < 8; ++v) acc[dt][v] *= alpha;
#pragma unroll
        for (int v = 0; v < 8; ++v) accl[v] *= alpha;

        // ---- l += 1 @ P^T ; O^T += V^T @ P^T ----
        accl = __builtin_amdgcn_wmma_f32_16x16x32_f16(
            false, ones, false, pb01, (short)0, accl, false, false);
        accl = __builtin_amdgcn_wmma_f32_16x16x32_f16(
            false, ones, false, pb23, (short)0, accl, false, false);
#pragma unroll
        for (int dt = 0; dt < 4; ++dt) {
            v16h va0 = load_frag(vrow[dt] + k0 + kg * 8, vrow[dt] + k0 + 16 + kg * 8);
            acc[dt] = __builtin_amdgcn_wmma_f32_16x16x32_f16(
                false, va0, false, pb01, (short)0, acc[dt], false, false);
            v16h va1 = load_frag(vrow[dt] + k0 + 32 + kg * 8, vrow[dt] + k0 + 48 + kg * 8);
            acc[dt] = __builtin_amdgcn_wmma_f32_16x16x32_f16(
                false, va1, false, pb23, (short)0, acc[dt], false, false);
        }
    };

    int k0 = 0;
    for (int j = 0; j < nblocks - 1; ++j, k0 += 64) process_block(k0, false);
    process_block(k0, true);  // only block that can touch the diagonal

    // ---- normalize (accl is per-lane) and store: contiguous b128 runs ----
    const float inv = 1.0f / accl[0];
    _Float16* yp = y + (size_t)(b * T_ + q) * C_ + h * D_ + 8 * kg;
#pragma unroll
    for (int dt = 0; dt < 4; ++dt) {
        v8h o;
#pragma unroll
        for (int v = 0; v < 8; ++v) o[v] = (_Float16)(acc[dt][v] * inv);
        *(v8h*)(yp + dt * 16) = o;
    }
}

// ---------------------------------------------------------------------------
// Host-side launch
// ---------------------------------------------------------------------------
static inline size_t align256(size_t x) { return (x + 255) & ~(size_t)255; }

extern "C" void kernel_launch(void* const* d_in, const int* in_sizes, int n_in,
                              void* d_out, int out_size, void* d_ws, size_t ws_size,
                              hipStream_t stream) {
    const float* x      = (const float*)d_in[0];  // [B,T,C]
    const float* W_attn = (const float*)d_in[1];  // [C,3C]
    const float* b_attn = (const float*)d_in[2];  // [3C]
    const float* W_proj = (const float*)d_in[3];  // [C,C]
    const float* b_proj = (const float*)d_in[4];  // [C]
    float* out = (float*)d_out;

    const long nX   = (long)B_ * T_ * C_;       // 8,388,608
    const long nWa  = (long)C_ * 3 * C_;        // 3,145,728
    const long nWp  = (long)C_ * C_;            // 1,048,576
    const long nQKV = (long)B_ * T_ * 3 * C_;   // 25,165,824

    char* ws = (char*)d_ws;
    size_t off = 0;
    _Float16* xh   = (_Float16*)(ws + off); off = align256(off + nX   * 2);
    _Float16* wat  = (_Float16*)(ws + off); off = align256(off + nWa  * 2);  // [3C][C]
    _Float16* wpt  = (_Float16*)(ws + off); off = align256(off + nWp  * 2);  // [C][C]
    _Float16* qkvh = (_Float16*)(ws + off); off = align256(off + nQKV * 2);
    _Float16* vth  = (_Float16*)(ws + off); off = align256(off + nX   * 2);  // [B,H,D,T]
    _Float16* yh   = (_Float16*)(ws + off); off = align256(off + nX   * 2);

    // 1) convert x to f16; transpose+convert weights to [N][K] f16
    f32_to_f16_kernel<<<(nX + 255) / 256, 256, 0, stream>>>(x, xh, nX);
    transpose_f32_to_f16_kernel<<<dim3(3 * C_ / 32, C_ / 32), 256, 0, stream>>>(
        W_attn, wat, C_, 3 * C_);
    transpose_f32_to_f16_kernel<<<dim3(C_ / 32, C_ / 32), 256, 0, stream>>>(
        W_proj, wpt, C_, C_);

    // 2) qkv = x @ W_attn + b_attn   (M=8192, N=3072, K=1024)
    gemm_wmma_kernel<true><<<dim3(3 * C_ / 128, (B_ * T_) / 128), 256, 0, stream>>>(
        xh, wat, b_attn, qkvh, nullptr, B_ * T_, 3 * C_, C_);

    // 3) extract V^T, then flash attention -> yh
    extract_vt_kernel<<<B_ * H_ * (T_ / 64), 256, 0, stream>>>(qkvh, vth);
    attn_kernel<<<(B_ * H_ * (T_ / 16)) / 8, 256, 0, stream>>>(qkvh, vth, yh);

    // 4) out = yh @ W_proj + b_proj  (M=8192, N=1024, K=1024), f32 output
    gemm_wmma_kernel<false><<<dim3(C_ / 128, (B_ * T_) / 128), 256, 0, stream>>>(
        yh, wpt, b_proj, nullptr, out, B_ * T_, C_, C_);
}
